// GruAtomic_49340584296573
// MI455X (gfx1250) — matrix operand order repaired and checked by reading, not compiled, not run on hardware
//
#include <hip/hip_runtime.h>

typedef __attribute__((ext_vector_type(16))) __bf16 v16bf;
typedef __attribute__((ext_vector_type(8)))  __bf16 v8bf;
typedef __attribute__((ext_vector_type(8)))  float  v8f;

#define GRU_B   256
#define GRU_T   512
#define GRU_IN  128
#define GRU_H   256
#define GRU_G   (3 * GRU_H)   // 768 gate rows
#define ROWS    16            // batch rows per workgroup (one WMMA M tile)
#define NTHREADS 512          // 16 waves

// workspace layout (bytes)
#define WS_WIH_OFF  0
#define WS_WHH_OFF  ((size_t)GRU_G * GRU_IN * 2)                 // 196608
#define WS_ZERO_OFF (WS_WHH_OFF + (size_t)GRU_G * GRU_H * 2)    // +393216

union AFrag { v8bf h[2]; v16bf v; };

// ---------------------------------------------------------------------------
// Prep: convert fp32 weights to bf16 in workspace; also write the zero word
// used to defeat loop-invariant hoisting of the streamed W_hh loads.
// ---------------------------------------------------------------------------
__global__ void __launch_bounds__(256) gru_prep_kernel(
    const float* __restrict__ wih_f, const float* __restrict__ whh_f,
    __bf16* __restrict__ wih_b, __bf16* __restrict__ whh_b,
    int* __restrict__ zero_w) {
  int i = blockIdx.x * 256 + threadIdx.x;
  if (i == 0) *zero_w = 0;
  if (i < GRU_G * GRU_IN) wih_b[i] = (__bf16)wih_f[i];
  if (i < GRU_G * GRU_H)  whh_b[i] = (__bf16)whh_f[i];
}

// B fragment (32x16 bf16): lane n=lane&15 selects gate row, lane group selects
// K half; 16 contiguous K elements per lane -> 32B load from W[row][k0..].
__device__ __forceinline__ v16bf load_b_tile(const __bf16* W, int ldk,
                                             int rowbase, int k0, int lane) {
  int n  = lane & 15;
  int hi = lane >> 4;
  return *(const v16bf*)(W + (size_t)(rowbase + n) * ldk + k0 + 16 * hi);
}

// ---------------------------------------------------------------------------
// Persistent GRU kernel: 16 blocks x 512 threads; each block owns 16 batch
// rows and runs the full T=512 recurrence privately (batch rows independent).
// W_ih fragments live in VGPRs for the whole run; W_hh streams from L2.
// ---------------------------------------------------------------------------
__global__ void __launch_bounds__(NTHREADS) gru_rnn_kernel(
    const float* __restrict__ x, const float* __restrict__ b_ih,
    const float* __restrict__ b_hh, const __bf16* __restrict__ wih,
    const __bf16* __restrict__ whh, const int* zoffp,
    float* __restrict__ out) {
  __shared__ __align__(32) __bf16 sh_x [ROWS * GRU_IN];  //  4 KB  x_t   (bf16)
  __shared__ __align__(32) __bf16 sh_hb[ROWS * GRU_H];   //  8 KB  h     (bf16)
  __shared__ __align__(32) float  sh_hf[ROWS * GRU_H];   // 16 KB  h     (fp32)

  const int tid  = threadIdx.x;
  const int wave = tid >> 5;
  const int lane = tid & 31;
  const int n    = lane & 15;   // WMMA N index / A-row M index
  const int hig  = lane >> 4;   // lane group
  const int kbA  = hig * 16;    // A-fragment K base inside a 32-chunk
  const int col0 = wave * 16;   // this wave's hidden-column block
  const int b0   = blockIdx.x * ROWS;

  // gate-row bases for this wave's r / z / n tiles
  const int gr = col0, gz = GRU_H + col0, gn = 2 * GRU_H + col0;

  // per-lane biases (t-invariant, in VGPRs)
  const float bir = b_ih[gr + n], bhr = b_hh[gr + n];
  const float biz = b_ih[gz + n], bhz = b_hh[gz + n];
  const float bin = b_ih[gn + n], bhn = b_hh[gn + n];

  // W_ih fragments held in registers for the whole recurrence: 12 x v16bf.
  v16bf wfr[4], wfz[4], wfn[4];
#pragma unroll
  for (int kk = 0; kk < 4; ++kk) {
    wfr[kk] = load_b_tile(wih, GRU_IN, gr, kk * 32, lane);
    wfz[kk] = load_b_tile(wih, GRU_IN, gz, kk * 32, lane);
    wfn[kk] = load_b_tile(wih, GRU_IN, gn, kk * 32, lane);
  }

  // h0 = 0
  for (int i = tid; i < ROWS * GRU_H; i += NTHREADS) {
    sh_hf[i] = 0.0f;
    sh_hb[i] = (__bf16)0.0f;
  }
  __syncthreads();

  for (int t = 0; t < GRU_T; ++t) {
    // volatile zero read -> W_hh base is loop-variant -> LICM cannot hoist
    // the streamed weight loads into (spilled) registers.
    const int zo = *(volatile const int*)zoffp;
    const __bf16* whh_t = whh + zo;

    // ---- stage x_t rows fp32 -> bf16 into LDS (coalesced) ----
    for (int i = tid; i < ROWS * GRU_IN; i += NTHREADS) {
      int r = i >> 7, c = i & (GRU_IN - 1);
      sh_x[i] = (__bf16)x[((size_t)(b0 + r) * GRU_T + t) * GRU_IN + c];
    }
    __syncthreads();

    v8f axr = {}, axz = {}, axn = {};   // x-projection accumulators
    v8f ahr = {}, ahz = {}, ahn = {};   // h-projection accumulators

    // ---- input projection: K = 128, weights from registers ----
#pragma unroll
    for (int kk = 0; kk < 4; ++kk) {
      AFrag a;
      const v8bf* pa = (const v8bf*)(sh_x + n * GRU_IN + kk * 32 + kbA);
      a.h[0] = pa[0];            // K = kbA .. kbA+7
      a.h[1] = pa[1];            // K = kbA+8 .. kbA+15
      axr = __builtin_amdgcn_wmma_f32_16x16x32_bf16(false, a.v, false, wfr[kk], (short)0, axr, false, false);
      axz = __builtin_amdgcn_wmma_f32_16x16x32_bf16(false, a.v, false, wfz[kk], (short)0, axz, false, false);
      axn = __builtin_amdgcn_wmma_f32_16x16x32_bf16(false, a.v, false, wfn[kk], (short)0, axn, false, false);
    }
    // ---- hidden projection: K = 256, weights streamed from L2 ----
#pragma unroll
    for (int k0 = 0; k0 < GRU_H; k0 += 32) {
      AFrag a;
      const v8bf* pa = (const v8bf*)(sh_hb + n * GRU_H + k0 + kbA);
      a.h[0] = pa[0];
      a.h[1] = pa[1];
      v16bf br_ = load_b_tile(whh_t, GRU_H, gr, k0, lane);
      v16bf bz_ = load_b_tile(whh_t, GRU_H, gz, k0, lane);
      v16bf bn_ = load_b_tile(whh_t, GRU_H, gn, k0, lane);
      ahr = __builtin_amdgcn_wmma_f32_16x16x32_bf16(false, a.v, false, br_, (short)0, ahr, false, false);
      ahz = __builtin_amdgcn_wmma_f32_16x16x32_bf16(false, a.v, false, bz_, (short)0, ahz, false, false);
      ahn = __builtin_amdgcn_wmma_f32_16x16x32_bf16(false, a.v, false, bn_, (short)0, ahn, false, false);
    }
    __syncthreads();   // all waves done reading sh_hb / sh_x

    // prefetch next step's x rows into cache (global_prefetch_b8)
    if (tid < ROWS && t + 1 < GRU_T) {
      __builtin_prefetch(&x[((size_t)(b0 + tid) * GRU_T + (t + 1)) * GRU_IN], 0, 0);
    }

    // ---- fused gate math, entirely in registers of the owning wave ----
    // C/D layout: element e, lane -> m = e + 8*hig, n = lane&15
#pragma unroll
    for (int e = 0; e < 8; ++e) {
      int m    = e + 8 * hig;
      int hidx = m * GRU_H + col0 + n;
      float hold = sh_hf[hidx];
      float rg = 1.0f / (1.0f + __expf(-(axr[e] + bir + ahr[e] + bhr)));
      float zg = 1.0f / (1.0f + __expf(-(axz[e] + biz + ahz[e] + bhz)));
      float ng = tanhf(axn[e] + bin + rg * (ahn[e] + bhn));
      float hnew = (1.0f - zg) * ng + zg * hold;
      sh_hf[hidx] = hnew;
      sh_hb[hidx] = (__bf16)hnew;
      out[((size_t)(b0 + m) * GRU_T + t) * GRU_H + col0 + n] = hnew;
    }
    __syncthreads();   // h_new visible before next step's GEMM / x overwrite
  }
}

// ---------------------------------------------------------------------------
extern "C" void kernel_launch(void* const* d_in, const int* in_sizes, int n_in,
                              void* d_out, int out_size, void* d_ws, size_t ws_size,
                              hipStream_t stream) {
  (void)in_sizes; (void)n_in; (void)out_size; (void)ws_size;
  const float* x    = (const float*)d_in[0];
  const float* Wih  = (const float*)d_in[1];
  const float* Whh  = (const float*)d_in[2];
  const float* bih  = (const float*)d_in[3];
  const float* bhh  = (const float*)d_in[4];
  float*       out  = (float*)d_out;

  __bf16* wih_b  = (__bf16*)((char*)d_ws + WS_WIH_OFF);   // 192 KB
  __bf16* whh_b  = (__bf16*)((char*)d_ws + WS_WHH_OFF);   // 384 KB
  int*    zero_w = (int*)((char*)d_ws + WS_ZERO_OFF);

  gru_prep_kernel<<<(GRU_G * GRU_H + 255) / 256, 256, 0, stream>>>(
      Wih, Whh, wih_b, whh_b, zero_w);
  gru_rnn_kernel<<<GRU_B / ROWS, NTHREADS, 0, stream>>>(
      x, bih, bhh, wih_b, whh_b, zero_w, out);
}
